// PostProcessCustomGrounding_36739150250376
// MI455X (gfx1250) — matrix operand order repaired and checked by reading, not compiled, not run on hardware
//
#include <hip/hip_runtime.h>
#include <hip/hip_bf16.h>

typedef __attribute__((ext_vector_type(16))) _Float16 v16h;
typedef __attribute__((ext_vector_type(8)))  _Float16 v8h;
typedef __attribute__((ext_vector_type(8)))  float    v8f;

#define BATCH 64
#define QDIM  900
#define TDIM  256
#define CDIM  91
#define CPAD  96
#define NSEL  300
#define PSTR  264           // LDS row stride in halves (256 + 8 pad, staggers banks)
#define NFLAT (QDIM * CPAD) // 86400 padded scores per batch

// Fast sigmoid: 1 + e^{-x} >= 1, so a raw v_rcp_f32 (~1 ulp) is safe; avoids
// the IEEE divide expansion (v_div_scale/v_div_fmas + Newton steps).
__device__ __forceinline__ float fast_sigmoid(float x) {
    return __builtin_amdgcn_rcpf(1.0f + __expf(-x));
}

// ---------------------------------------------------------------------------
// Kernel 1: scores[b,q,c] = sum_t sigmoid(logits[b,q,t]) * pmap[c,t]
// f16 WMMA (16x16x32), f32 accumulate. Padding columns (c>=91) written as -1.
// ---------------------------------------------------------------------------
__global__ void __launch_bounds__(256)
grounding_gemm_sigmoid_wmma(const float* __restrict__ logits,
                            const float* __restrict__ pmap,
                            float* __restrict__ scores) {
    __shared__ _Float16 Plds[CPAD * PSTR];

    const int b     = blockIdx.x;
    const int qtile = blockIdx.y;
    const int tid   = threadIdx.x;

    // Cooperative load: positive_map -> f16 LDS, rows 91..95 zeroed.
    for (int i = tid; i < CPAD * TDIM; i += 256) {
        const int c = i >> 8;        // i / 256
        const int t = i & 255;       // i % 256
        const float v = (c < CDIM) ? pmap[c * TDIM + t] : 0.0f;
        Plds[c * PSTR + t] = (_Float16)v;
    }
    __syncthreads();

    const int wave  = tid >> 5;
    const int lane  = tid & 31;
    const int lhalf = lane >> 4;     // 0: lanes 0-15, 1: lanes 16-31
    const int lrow  = lane & 15;     // A-matrix row owned by this lane
    const int col   = lane & 15;     // B/C/D column owned by this lane

    const int q0 = qtile * 128 + wave * 16;
    if (q0 >= QDIM) return;          // after the barrier: safe to exit

    const int  qrow     = q0 + lrow;
    const bool rowValid = (qrow < QDIM);
    const float* rowBase =
        logits + ((size_t)b * QDIM + (rowValid ? qrow : 0)) * TDIM;

    v8f acc[6] = {};

    for (int kk = 0; kk < TDIM; kk += 32) {
        // ---- Prefetch all 6 B fragments first so the 12 ds_load_b128s are
        //      in flight before any WMMA needs them (graduated s_wait_dscnt).
        v16h bf[6];
#pragma unroll
        for (int nt = 0; nt < 6; ++nt) {
            // B fragment 32x16: lane (half h) col n, element e: K = kk + (e<8 ? 0 : 16) + h*8 + (e&7)
            const int off = (nt * 16 + col) * PSTR + kk + lhalf * 8;
            const v8h blo = *(const v8h*)&Plds[off];
            const v8h bhi = *(const v8h*)&Plds[off + 16];
            bf[nt] = __builtin_shufflevector(
                blo, bhi, 0, 1, 2, 3, 4, 5, 6, 7, 8, 9, 10, 11, 12, 13, 14, 15);
        }

        // ---- A fragment: 16x32 f16, per documented wave32 layout ----
        // lane (half h) element e: K = kk + (e<8 ? 0 : 16) + h*8 + (e&7)
        v16h a;
        if (rowValid) {
            const float* p = rowBase + kk + lhalf * 8;
            const float4 f0 = *(const float4*)(p);
            const float4 f1 = *(const float4*)(p + 4);
            const float4 f2 = *(const float4*)(p + 16);
            const float4 f3 = *(const float4*)(p + 20);
            const float t16[16] = {f0.x, f0.y, f0.z, f0.w, f1.x, f1.y, f1.z, f1.w,
                                   f2.x, f2.y, f2.z, f2.w, f3.x, f3.y, f3.z, f3.w};
#pragma unroll
            for (int e = 0; e < 16; ++e)
                a[e] = (_Float16)fast_sigmoid(t16[e]);
        } else {
#pragma unroll
            for (int e = 0; e < 16; ++e) a[e] = (_Float16)0.0f;
        }

        // ---- 6 N-tiles of 16 classes each ----
#pragma unroll
        for (int nt = 0; nt < 6; ++nt) {
            acc[nt] = __builtin_amdgcn_wmma_f32_16x16x32_f16(
                false, a, false, bf[nt], (short)0, acc[nt], false, false);
        }
    }

    // ---- Store D: lane holds col, VGPR r -> row m = r + lhalf*8 ----
    float* S = scores + (size_t)b * QDIM * CPAD;
#pragma unroll
    for (int nt = 0; nt < 6; ++nt) {
#pragma unroll
        for (int r = 0; r < 8; ++r) {
            const int m = r + lhalf * 8;
            const int q = q0 + m;
            const int c = nt * 16 + col;
            if (q < QDIM)
                S[(size_t)q * CPAD + c] = (c < CDIM) ? acc[nt][r] : -1.0f;
        }
    }
}

// ---------------------------------------------------------------------------
// Kernel 2: per-batch top-300 (lowest-index tie-break, like jax.lax.top_k),
// then label/box decode + scaling. One 1024-thread block per batch; scores
// are cached privately so the 300 extraction rounds never touch HBM.
// ---------------------------------------------------------------------------
__global__ void __launch_bounds__(1024)
grounding_topk_decode(const float* __restrict__ scores,
                      const float* __restrict__ boxes,
                      const int* __restrict__ tsizes,
                      float* __restrict__ out) {
    const int b   = blockIdx.x;
    const int tid = threadIdx.x;
    const int NT  = 1024;

    __shared__ float rv[1024];
    __shared__ int   ri[1024];
    __shared__ float resv[NSEL];
    __shared__ int   resq[NSEL];
    __shared__ int   resc[NSEL];
    __shared__ int   winner;

    const float* S = scores + (size_t)b * NFLAT;
    float lv[85];
    int cnt = 0;
    for (int i = tid; i < NFLAT; i += NT) lv[cnt++] = S[i];

    for (int k = 0; k < NSEL; ++k) {
        // thread-local argmax (strict > keeps the smallest flat index)
        float bv = -3.0e38f;
        int   bi = 0x7FFFFFFF;
        for (int j = 0; j < cnt; ++j) {
            if (lv[j] > bv) { bv = lv[j]; bi = tid + j * NT; }
        }
        rv[tid] = bv;
        ri[tid] = bi;
        __syncthreads();
        for (int s = 512; s > 0; s >>= 1) {
            if (tid < s) {
                const float ov = rv[tid + s];
                const int   oi = ri[tid + s];
                if (ov > rv[tid] || (ov == rv[tid] && oi < ri[tid])) {
                    rv[tid] = ov;
                    ri[tid] = oi;
                }
            }
            __syncthreads();
        }
        if (tid == 0) {
            const int gi = ri[0];
            resv[k] = rv[0];
            resq[k] = gi / CPAD;
            resc[k] = gi % CPAD;
            winner  = gi;
        }
        __syncthreads();
        const int gi = winner;
        if ((gi & (NT - 1)) == tid) lv[gi >> 10] = -3.0e38f; // owner invalidates
        __syncthreads();
    }

    // Output layout: scores[64*300] | labels[64*300] | boxes[64*300*4]
    if (tid < NSEL) {
        const int k = tid;
        const int q = resq[k];
        const int c = resc[k];
        out[b * NSEL + k]                 = resv[k];
        out[BATCH * NSEL + b * NSEL + k]  = (float)c;

        const float* bx = boxes + ((size_t)b * QDIM + q) * 4;
        const float cx = bx[0], cy = bx[1], w = bx[2], h = bx[3];
        const float ih = (float)tsizes[b * 2 + 0];
        const float iw = (float)tsizes[b * 2 + 1];
        float* ob = out + 2 * BATCH * NSEL + ((size_t)(b * NSEL + k)) * 4;
        ob[0] = (cx - 0.5f * w) * iw;
        ob[1] = (cy - 0.5f * h) * ih;
        ob[2] = (cx + 0.5f * w) * iw;
        ob[3] = (cy + 0.5f * h) * ih;
    }
}

// ---------------------------------------------------------------------------
extern "C" void kernel_launch(void* const* d_in, const int* in_sizes, int n_in,
                              void* d_out, int out_size, void* d_ws, size_t ws_size,
                              hipStream_t stream) {
    (void)in_sizes; (void)n_in; (void)out_size; (void)ws_size;

    const float* logits = (const float*)d_in[0]; // (64, 900, 256) f32
    const float* boxes  = (const float*)d_in[1]; // (64, 900, 4)   f32
    const float* pmap   = (const float*)d_in[2]; // (91, 256)      f32
    const int*   tsz    = (const int*)d_in[3];   // (64, 2)        i32
    // d_in[4] = num_select (300), fixed by the reference setup.

    float* ws  = (float*)d_ws;   // needs 64*900*96*4 = 22.1 MB, rewritten each call
    float* out = (float*)d_out;  // 115200 f32: scores | labels | boxes

    grounding_gemm_sigmoid_wmma<<<dim3(BATCH, 8), 256, 0, stream>>>(logits, pmap, ws);
    grounding_topk_decode<<<dim3(BATCH), 1024, 0, stream>>>(ws, boxes, tsz, out);
}
